// LaguerreKANLayer_10720238371160
// MI455X (gfx1250) — matrix operand order repaired and checked by reading, not compiled, not run on hardware
//
#include <hip/hip_runtime.h>
#include <hip/hip_bf16.h>

// LaguerreKAN layer for MI455X (gfx1250):
//   Phase 1: basis  A[b, i*8+d] = Laguerre_d(tanh(x[b,i]))  (f16, [8192][4096])
//   Phase 2: weight Wt[o, i*8+d] = coeffs[i,o,d]            (f16, [512][4096])
//   Phase 3: Y = A * Wt^T via v_wmma_f32_16x16x32_f16, f32 accumulate.
//            128x128 block tile, 8 waves, each wave = 32x64 register tile
//            (2 A-frags x 4 B-frags -> 8 WMMAs per 12 ds_load_b128).

typedef _Float16 v16h __attribute__((ext_vector_type(16)));
typedef _Float16 v8h  __attribute__((ext_vector_type(8)));
typedef float    v8f  __attribute__((ext_vector_type(8)));

#define INPUT_DIM  512
#define OUTPUT_DIM 512
#define NDEG       8                    // DEGREE+1
#define KDIM       (INPUT_DIM * NDEG)   // 4096
#define BATCH      8192

#define BM 128
#define BN 128
#define BK 32
#define LDT 40   // padded LDS row stride in halves (32 + 8) -> conflict-free b128

// ---------------- Phase 1: Laguerre basis (f32 -> f16) ----------------
__global__ void laguerre_basis_kernel(const float* __restrict__ x,
                                      _Float16* __restrict__ A,
                                      int total) {
    int idx = blockIdx.x * blockDim.x + threadIdx.x;
    if (idx >= total) return;
    float xv = tanhf(x[idx]);
    float L[NDEG];
    L[0] = 1.0f;
    L[1] = 1.5f - xv;   // 1 + alpha - x, alpha = 0.5
#pragma unroll
    for (int k = 2; k < NDEG; ++k) {
        // k*L_k = (2(k-1)+1+alpha - x)*L_{k-1} - (k-1+alpha)*L_{k-2}
        L[k] = ((2.0f * (k - 1) + 1.5f - xv) * L[k - 1]
                - ((k - 1) + 0.5f) * L[k - 2]) * (1.0f / k);
    }
    v8h out;
#pragma unroll
    for (int d = 0; d < NDEG; ++d) out[d] = (_Float16)L[d];
    *(v8h*)(A + (size_t)idx * NDEG) = out;   // global_store_b128
}

// ---------------- Phase 2: coeffs [i][o][d] f32 -> Wt [o][i*8+d] f16 ----------------
__global__ void coeffs_pack_kernel(const float* __restrict__ c,
                                   _Float16* __restrict__ Wt,
                                   int total) {
    int idx = blockIdx.x * blockDim.x + threadIdx.x;  // idx = o*INPUT_DIM + i, i fastest
    if (idx >= total) return;
    int i = idx % INPUT_DIM;
    int o = idx / INPUT_DIM;
    const float* src = c + (((size_t)i * OUTPUT_DIM + o) * NDEG); // contiguous 8 f32
    v8h out;
#pragma unroll
    for (int d = 0; d < NDEG; ++d) out[d] = (_Float16)src[d];
    *(v8h*)(Wt + (size_t)o * KDIM + (size_t)i * NDEG) = out;
}

// ---------------- Phase 3: WMMA GEMM  Y[8192][512] = A[8192][4096] x Wt^T ----------------
__global__ __launch_bounds__(256)
void wmma_gemm_kernel(const _Float16* __restrict__ A,
                      const _Float16* __restrict__ Wt,
                      float* __restrict__ Y) {
    __shared__ __attribute__((aligned(16))) _Float16 As[BM * LDT];
    __shared__ __attribute__((aligned(16))) _Float16 Bs[BN * LDT];

    const int tid  = threadIdx.x;
    const int wave = tid >> 5;        // 0..7
    const int lane = tid & 31;
    const int half = lane >> 4;       // 0 => lanes 0-15, 1 => lanes 16-31
    const int lrow = lane & 15;

    const int wm = wave & 3;          // row quadrant: 4 waves x 32 rows
    const int wn = wave >> 2;         // col half:     2 waves x 64 cols

    const int m0 = blockIdx.x * BM;
    const int n0 = blockIdx.y * BN;

    v8f acc[2][4];
#pragma unroll
    for (int mi = 0; mi < 2; ++mi)
#pragma unroll
        for (int j = 0; j < 4; ++j) acc[mi][j] = (v8f)(0.0f);

    for (int k0 = 0; k0 < KDIM; k0 += BK) {
        // ---- stage 128x32 A-tile and 128x32 Wt-tile into LDS (16B chunks) ----
#pragma unroll
        for (int p = 0; p < 2; ++p) {
            int c16 = tid + p * 256;      // 512 chunks of 16B per tile
            int row = c16 >> 2;
            int cc  = c16 & 3;
            *(v8h*)&As[row * LDT + cc * 8] =
                *(const v8h*)&A[(size_t)(m0 + row) * KDIM + k0 + cc * 8];
            *(v8h*)&Bs[row * LDT + cc * 8] =
                *(const v8h*)&Wt[(size_t)(n0 + row) * KDIM + k0 + cc * 8];
        }
        __syncthreads();

        // ---- load ALL fragments into distinct registers first ----
        // A fragment (16x32 f16, ISA 7.12.2): lane holds row M=lrow;
        // lanes 0-15: K 0-7 & 16-23, lanes 16-31: K 8-15 & 24-31.
        v16h afrag[2];
#pragma unroll
        for (int mi = 0; mi < 2; ++mi) {
            const _Float16* ap = &As[(wm * 32 + mi * 16 + lrow) * LDT + half * 8];
            v8h a_lo = *(const v8h*)ap;          // K = half*8 .. +7
            v8h a_hi = *(const v8h*)(ap + 16);   // K = 16+half*8 .. +7
            afrag[mi] = __builtin_shufflevector(a_lo, a_hi,
                0,1,2,3,4,5,6,7,8,9,10,11,12,13,14,15);
        }
        // B fragment (32x16 f16): lane holds column N=lrow;
        // lanes 0-15: K 0-7 & 16-23, lanes 16-31: K 8-15 & 24-31 (same as A).
        v16h bfrag[4];
#pragma unroll
        for (int j = 0; j < 4; ++j) {
            const _Float16* bp = &Bs[(wn * 64 + j * 16 + lrow) * LDT + half * 8];
            v8h b_lo = *(const v8h*)bp;
            v8h b_hi = *(const v8h*)(bp + 16);
            bfrag[j] = __builtin_shufflevector(b_lo, b_hi,
                0,1,2,3,4,5,6,7,8,9,10,11,12,13,14,15);
        }

        // ---- 8 back-to-back WMMAs, each A reused 4x, each B reused 2x ----
#pragma unroll
        for (int mi = 0; mi < 2; ++mi)
#pragma unroll
            for (int j = 0; j < 4; ++j)
                acc[mi][j] = __builtin_amdgcn_wmma_f32_16x16x32_f16(
                    false, afrag[mi], false, bfrag[j], (short)0, acc[mi][j],
                    false, false);
        __syncthreads();
    }

    // ---- write back: D layout = 8 f32 VGPRs; M = half*8 + v, N = lrow ----
#pragma unroll
    for (int mi = 0; mi < 2; ++mi) {
        const int orow = m0 + wm * 32 + mi * 16 + half * 8;
#pragma unroll
        for (int j = 0; j < 4; ++j) {
            const int col = n0 + wn * 64 + j * 16 + lrow;
#pragma unroll
            for (int v = 0; v < 8; ++v) {
                Y[(size_t)(orow + v) * OUTPUT_DIM + col] = acc[mi][j][v];
            }
        }
    }
}

extern "C" void kernel_launch(void* const* d_in, const int* in_sizes, int n_in,
                              void* d_out, int out_size, void* d_ws, size_t ws_size,
                              hipStream_t stream) {
    const float* x = (const float*)d_in[0];                 // [8192][512] f32
    const float* coeffs = (const float*)d_in[1];            // [512][512][8] f32
    float* y = (float*)d_out;                               // [8192][512] f32

    const int batch = in_sizes[0] / INPUT_DIM;              // 8192

    _Float16* Abuf = (_Float16*)d_ws;                       // [batch][4096] f16 (64 MB)
    _Float16* Wbuf = Abuf + (size_t)batch * KDIM;           // [512][4096]  f16 (4 MB)

    {
        int total = batch * INPUT_DIM;
        laguerre_basis_kernel<<<(total + 255) / 256, 256, 0, stream>>>(x, Abuf, total);
    }
    {
        int total = OUTPUT_DIM * INPUT_DIM;
        coeffs_pack_kernel<<<(total + 255) / 256, 256, 0, stream>>>(coeffs, Wbuf, total);
    }
    {
        dim3 grid(batch / BM, OUTPUT_DIM / BN);             // 64 x 4
        wmma_gemm_kernel<<<grid, 256, 0, stream>>>(Abuf, Wbuf, y);
    }
}
